// GCN_60224031425188
// MI455X (gfx1250) — matrix operand order repaired and checked by reading, not compiled, not run on hardware
//
#include <hip/hip_runtime.h>
#include <hip/hip_bf16.h>
#include <cstddef>

#define N_NODES 8192
#define FD 768
#define HIDD 384
#define OUTD 10
#define NNZE (N_NODES * 16)
#define DCATD (FD + 2 * HIDD) /* 1536 */

typedef __bf16 bf16x16 __attribute__((ext_vector_type(16)));
typedef __bf16 bf16x8  __attribute__((ext_vector_type(8)));
typedef __bf16 bf16x4  __attribute__((ext_vector_type(4)));
typedef float  f32x8   __attribute__((ext_vector_type(8)));

union ABf16 { bf16x16 v; bf16x8 h[2]; };

// monotone float<->uint encode for atomicMax over signed floats
__device__ __forceinline__ unsigned fenc(float f) {
  unsigned u = __float_as_uint(f);
  return (u & 0x80000000u) ? ~u : (u | 0x80000000u);
}
__device__ __forceinline__ float fdec(unsigned u) {
  return (u & 0x80000000u) ? __uint_as_float(u & 0x7FFFFFFFu) : __uint_as_float(~u);
}

__device__ __forceinline__ void cvt_store4(__bf16* dst, float4 v) {
  bf16x4 p;
  p.x = (__bf16)v.x; p.y = (__bf16)v.y; p.z = (__bf16)v.z; p.w = (__bf16)v.w;
  *(bf16x4*)dst = p;   // 2x v_cvt_pk_bf16_f32 + 8-byte ds_store
}

// -------- register-staged tiles (split load / store so loads stay in flight) --------
struct RegsA { float4 t[4]; };   // 128x32 A tile, 16 f32 per thread
struct RegsB { float4 t[2]; };   // 64x32 B tile, 8 f32 per thread

// A tile: rows m0..m0+127 of A[M,K] row-major, cols kt..kt+31
__device__ __forceinline__ void load_A128(const float* __restrict__ A, int K, int m0, int kt,
                                          RegsA& ra) {
#pragma unroll
  for (int j = 0; j < 4; ++j) {
    int idx = ((int)threadIdx.x + j * 256) * 4;
    int r = idx >> 5, k = idx & 31;
    ra.t[j] = *(const float4*)(A + (size_t)(m0 + r) * K + kt + k);
  }
}
__device__ __forceinline__ void store_A128(const RegsA& ra, __bf16* Al) {
#pragma unroll
  for (int j = 0; j < 4; ++j) {
    int idx = ((int)threadIdx.x + j * 256) * 4;
    int r = idx >> 5, k = idx & 31;
    cvt_store4(Al + r * 32 + k, ra.t[j]);
  }
}
// B^T source: B[Nc,K] row-major, tile rows n0..n0+63 -> Bl[n][k]
__device__ __forceinline__ void load_Bt(const float* __restrict__ B, int K, int n0, int kt,
                                        RegsB& rb) {
#pragma unroll
  for (int j = 0; j < 2; ++j) {
    int idx = ((int)threadIdx.x + j * 256) * 4;
    int r = idx >> 5, k = idx & 31;
    rb.t[j] = *(const float4*)(B + (size_t)(n0 + r) * K + kt + k);
  }
}
__device__ __forceinline__ void store_Bt(const RegsB& rb, __bf16* Bl) {
#pragma unroll
  for (int j = 0; j < 2; ++j) {
    int idx = ((int)threadIdx.x + j * 256) * 4;
    int r = idx >> 5, k = idx & 31;
    cvt_store4(Bl + r * 32 + k, rb.t[j]);
  }
}
// B source: B[K,Nc] row-major, tile (kt..kt+31) x (n0..n0+63) -> Bl[n][k] (transposed store)
__device__ __forceinline__ void load_Bn(const float* __restrict__ B, int Nc, int n0, int kt,
                                        RegsB& rb) {
#pragma unroll
  for (int j = 0; j < 2; ++j) {
    int idx = ((int)threadIdx.x + j * 256) * 4;
    int k = idx >> 6, n = idx & 63;
    rb.t[j] = *(const float4*)(B + (size_t)(kt + k) * Nc + n0 + n);
  }
}
__device__ __forceinline__ void store_Bn(const RegsB& rb, __bf16* Bl) {
#pragma unroll
  for (int j = 0; j < 2; ++j) {
    int idx = ((int)threadIdx.x + j * 256) * 4;
    int k = idx >> 6, n = idx & 63;
    Bl[(n + 0) * 32 + k] = (__bf16)rb.t[j].x;
    Bl[(n + 1) * 32 + k] = (__bf16)rb.t[j].y;
    Bl[(n + 2) * 32 + k] = (__bf16)rb.t[j].z;
    Bl[(n + 3) * 32 + k] = (__bf16)rb.t[j].w;
  }
}

// per-wave: A frag (16x32) at row block mw, 4 B frags covering 64 cols -> 4 WMMAs.
// All fragment ds_loads are issued before the first WMMA so the compiler can
// clause them and use partial s_wait_dscnt drains.
__device__ __forceinline__ void mma_step4(const __bf16* Al, const __bf16* Bl,
                                          int mw, int lane, f32x8* acc) {
  const int half = lane >> 4, ln = lane & 15;
  ABf16 fa, fb[4];
  const int m = mw + ln;
  // A layout (16-bit, 16x32): lane half 0 -> K 0..7 & 16..23; half 1 -> K 8..15 & 24..31
  fa.h[0] = *(const bf16x8*)(Al + m * 32 + half * 8);
  fa.h[1] = *(const bf16x8*)(Al + m * 32 + 16 + half * 8);
#pragma unroll
  for (int nf = 0; nf < 4; ++nf) {
    const int n = nf * 16 + ln;
    // B layout (16-bit, 32x16): lane half 0 -> K 0..7 & 16..23; half 1 -> K 8..15 & 24..31
    fb[nf].h[0] = *(const bf16x8*)(Bl + n * 32 + half * 8);
    fb[nf].h[1] = *(const bf16x8*)(Bl + n * 32 + 16 + half * 8);
  }
#pragma unroll
  for (int nf = 0; nf < 4; ++nf) {
    acc[nf] = __builtin_amdgcn_wmma_f32_16x16x32_bf16(false, fa.v, false, fb[nf].v, (short)0,
                                                      acc[nf], false, false);
  }
}

// ---------------- generic WMMA GEMM:  C = act(A @ op(B) + bias) ----------------
// transB=1: B is [Nc,K] (computes A@B^T). transB=0: B is [K,Nc].
// epi: 0 = raw, 1 = +bias then lrelu(0.01), 2 = +bias
// requires M%128==0, Nc%64==0, K%32==0
__global__ __launch_bounds__(256)
void k_wmma_gemm(const float* __restrict__ A, const float* __restrict__ B,
                 float* __restrict__ C, int M, int Nc, int K,
                 const float* __restrict__ bias, int epi, int transB) {
  __shared__ __bf16 Al[2][128 * 32];
  __shared__ __bf16 Bl[2][64 * 32];
  const int n0 = blockIdx.x * 64;
  const int m0 = blockIdx.y * 128;
  const int lane = threadIdx.x & 31;
  const int wave = threadIdx.x >> 5;
  const int mw = wave * 16;
  f32x8 acc[4];
#pragma unroll
  for (int nf = 0; nf < 4; ++nf)
#pragma unroll
    for (int i = 0; i < 8; ++i) acc[nf][i] = 0.f;

  RegsA ra; RegsB rb;
  load_A128(A, K, m0, 0, ra);
  if (transB) load_Bt(B, K, n0, 0, rb);
  else        load_Bn(B, Nc, n0, 0, rb);
  store_A128(ra, Al[0]);
  if (transB) store_Bt(rb, Bl[0]);
  else        store_Bn(rb, Bl[0]);
  __syncthreads();

  const int S = K >> 5;
  for (int s = 0; s < S; ++s) {
    const int cur = s & 1, nxt = cur ^ 1;
    const bool more = (s + 1 < S);
    if (more) {                                   // issue next-tile global loads...
      load_A128(A, K, m0, (s + 1) * 32, ra);
      if (transB) load_Bt(B, K, n0, (s + 1) * 32, rb);
      else        load_Bn(B, Nc, n0, (s + 1) * 32, rb);
    }
    mma_step4(Al[cur], Bl[cur], mw, lane, acc);   // ...hidden under the WMMAs
    if (more) {
      store_A128(ra, Al[nxt]);
      if (transB) store_Bt(rb, Bl[nxt]);
      else        store_Bn(rb, Bl[nxt]);
    }
    __syncthreads();
  }
  const int half = lane >> 4, ln = lane & 15;
#pragma unroll
  for (int nf = 0; nf < 4; ++nf) {
    const int ncol = n0 + nf * 16 + ln;
    const float bv = (epi != 0) ? bias[ncol] : 0.f;
#pragma unroll
    for (int v = 0; v < 8; ++v) {
      const int mrow = m0 + mw + half * 8 + v;   // C layout: VGPR v -> M = v + 8*half
      float val = acc[nf][v] + bv;
      if (epi == 1) val = val >= 0.f ? val : 0.01f * val;
      C[(size_t)mrow * Nc + ncol] = val;
    }
  }
}

// --------- attn GEMM with fused relu*W2 reduction: score[d] += sum_j relu((W1@cat)[j,d]+b1[j])*W2[j]
__global__ __launch_bounds__(256)
void k_attn_gemm_score(const float* __restrict__ W1, const float* __restrict__ cat,
                       const float* __restrict__ b1, const float* __restrict__ W2,
                       float* __restrict__ score) {
  __shared__ __bf16 Al[2][128 * 32];
  __shared__ __bf16 Bl[2][64 * 32];
  const int n0 = blockIdx.x * 64;    // over DCAT
  const int m0 = blockIdx.y * 128;   // over j
  const int lane = threadIdx.x & 31;
  const int wave = threadIdx.x >> 5;
  const int mw = wave * 16;
  f32x8 acc[4];
#pragma unroll
  for (int nf = 0; nf < 4; ++nf)
#pragma unroll
    for (int i = 0; i < 8; ++i) acc[nf][i] = 0.f;

  RegsA ra; RegsB rb;
  load_A128(W1, N_NODES, m0, 0, ra);
  load_Bn(cat, DCATD, n0, 0, rb);
  store_A128(ra, Al[0]);
  store_Bn(rb, Bl[0]);
  __syncthreads();

  const int S = N_NODES >> 5;
  for (int s = 0; s < S; ++s) {
    const int cur = s & 1, nxt = cur ^ 1;
    const bool more = (s + 1 < S);
    if (more) {
      load_A128(W1, N_NODES, m0, (s + 1) * 32, ra);
      load_Bn(cat, DCATD, n0, (s + 1) * 32, rb);
    }
    mma_step4(Al[cur], Bl[cur], mw, lane, acc);
    if (more) {
      store_A128(ra, Al[nxt]);
      store_Bn(rb, Bl[nxt]);
    }
    __syncthreads();
  }
  const int half = lane >> 4, ln = lane & 15;
  float p[4] = {0.f, 0.f, 0.f, 0.f};
#pragma unroll
  for (int v = 0; v < 8; ++v) {
    const int j = m0 + mw + half * 8 + v;
    const float bb = b1[j], ww = W2[j];
#pragma unroll
    for (int nf = 0; nf < 4; ++nf) {
      float z = acc[nf][v] + bb;
      p[nf] += (z > 0.f ? z : 0.f) * ww;
    }
  }
#pragma unroll
  for (int nf = 0; nf < 4; ++nf) {
    float q = p[nf] + __shfl_xor(p[nf], 16, 32);  // combine M halves
    if (half == 0) atomicAdd(&score[n0 + nf * 16 + ln], q);
  }
}

// ---------------- GraphNorm ----------------
__global__ __launch_bounds__(256)
void k_colstats(const float* __restrict__ X, const float* __restrict__ w,
                const float* __restrict__ b, const float* __restrict__ ms,
                float* __restrict__ scale, float* __restrict__ shift) {
  const int c = blockIdx.x;
  const int t = threadIdx.x;
  float s1 = 0.f, s2 = 0.f;
  for (int i = t; i < N_NODES; i += 256) {
    float v = X[(size_t)i * FD + c];
    s1 += v; s2 += v * v;
  }
  __shared__ float r1[256], r2[256];
  r1[t] = s1; r2[t] = s2; __syncthreads();
  for (int s = 128; s > 0; s >>= 1) {
    if (t < s) { r1[t] += r1[t + s]; r2[t] += r2[t + s]; }
    __syncthreads();
  }
  if (t == 0) {
    float m = r1[0] / (float)N_NODES;
    float ex2 = r2[0] / (float)N_NODES;
    float msv = ms[c];
    float var = ex2 - 2.f * msv * m * m + msv * msv * m * m;  // E[(x-ms*m)^2]
    float inv = w[c] * rsqrtf(var + 1e-5f);
    scale[c] = inv;
    shift[c] = b[c] - inv * msv * m;
  }
}

__global__ __launch_bounds__(256)
void k_gn_apply(const float* __restrict__ x, const float* __restrict__ scale,
                const float* __restrict__ shift, float* __restrict__ y, size_t n) {
  size_t i4 = ((size_t)blockIdx.x * 256 + threadIdx.x) * 4;
  if (i4 >= n) return;
  float4 v = *(const float4*)(x + i4);
  int c = (int)(i4 % FD);
  float4 o;
  o.x = scale[c + 0] * v.x + shift[c + 0];
  o.y = scale[c + 1] * v.y + shift[c + 1];
  o.z = scale[c + 2] * v.z + shift[c + 2];
  o.w = scale[c + 3] * v.w + shift[c + 3];
  *(float4*)(y + i4) = o;
}

// ---------------- utility ----------------
__global__ __launch_bounds__(256)
void k_fill_u32(unsigned* __restrict__ p, unsigned v, size_t n) {
  size_t i = (size_t)blockIdx.x * 256 + threadIdx.x;
  if (i < n) p[i] = v;
}

__global__ __launch_bounds__(256)
void k_rowdot(const float* __restrict__ XT, const float* __restrict__ ET,
              const float* __restrict__ att, float* __restrict__ sx, float* __restrict__ se) {
  const int i = blockIdx.x;
  const int t = threadIdx.x;
  float a0 = 0.f, a1 = 0.f;
  for (int f = t; f < FD; f += 256) {
    a0 += XT[(size_t)i * FD + f] * att[f];
    a1 += ET[(size_t)i * FD + f] * att[FD + f];
  }
  __shared__ float r0[256], r1[256];
  r0[t] = a0; r1[t] = a1; __syncthreads();
  for (int s = 128; s > 0; s >>= 1) {
    if (t < s) { r0[t] += r0[t + s]; r1[t] += r1[t + s]; }
    __syncthreads();
  }
  if (t == 0) { sx[i] = r0[0]; se[i] = r1[0]; }
}

// ---------------- edge (segment) kernels ----------------
__global__ __launch_bounds__(256)
void k_edge_alpha_max(const int* __restrict__ row, const int* __restrict__ col,
                      const float* __restrict__ sx, const float* __restrict__ se,
                      float* __restrict__ alpha, unsigned* __restrict__ mcol,
                      float* __restrict__ degE, float* __restrict__ degN) {
  int e = blockIdx.x * 256 + threadIdx.x;
  if (e >= NNZE) return;
  int r = row[e], c = col[e];
  float a = sx[r] + se[c];
  a = a >= 0.f ? a : 0.2f * a;
  alpha[e] = a;
  atomicMax(&mcol[c], fenc(a));
  atomicAdd(&degE[c], 1.f);
  atomicAdd(&degN[r], 1.f);
}

__global__ __launch_bounds__(256)
void k_edge_exp_sum(const int* __restrict__ col, const float* __restrict__ alpha,
                    const unsigned* __restrict__ mcol, float* __restrict__ ev,
                    float* __restrict__ ssum) {
  int e = blockIdx.x * 256 + threadIdx.x;
  if (e >= NNZE) return;
  int c = col[e];
  float x = __expf(alpha[e] - fdec(mcol[c]));
  ev[e] = x;
  atomicAdd(&ssum[c], x);
}

__global__ __launch_bounds__(256)
void k_edge_scatter_efeat(const int* __restrict__ row, const int* __restrict__ col,
                          const float* __restrict__ ev, const float* __restrict__ ssum,
                          const float* __restrict__ degE, const float* __restrict__ XT,
                          float* __restrict__ EF) {
  const int e = blockIdx.x;
  const int r = row[e], c = col[e];
  const float dE = degE[c];
  const float Bn = dE > 0.f ? 1.f / dE : 0.f;
  const float asft = ev[e] / (ssum[c] + 1e-16f);
  const float coef = Bn * asft;
  for (int f = threadIdx.x; f < FD; f += 256)
    atomicAdd(&EF[(size_t)c * FD + f], coef * XT[(size_t)r * FD + f]);
}

__global__ __launch_bounds__(256)
void k_edge_scatter_out(const int* __restrict__ row, const int* __restrict__ col,
                        const float* __restrict__ ev, const float* __restrict__ ssum,
                        const float* __restrict__ degN, const float* __restrict__ EF,
                        float* __restrict__ H) {
  const int e = blockIdx.x;
  const int r = row[e], c = col[e];
  const float dN = degN[r];
  const float Dn = dN > 0.f ? 1.f / dN : 0.f;
  const float asft = ev[e] / (ssum[c] + 1e-16f);
  const float coef = Dn * asft;
  for (int f = threadIdx.x; f < FD; f += 256)
    atomicAdd(&H[(size_t)r * FD + f], coef * EF[(size_t)c * FD + f]);
}

__global__ __launch_bounds__(256)
void k_bias_lrelu(float* __restrict__ h, const float* __restrict__ b, int ncols, size_t n,
                  float slope) {
  size_t i = (size_t)blockIdx.x * 256 + threadIdx.x;
  if (i >= n) return;
  float v = h[i] + b[(int)(i % ncols)];
  h[i] = v >= 0.f ? v : slope * v;
}

__global__ __launch_bounds__(256)
void k_concat(const float* __restrict__ x, const float* __restrict__ o1,
              const float* __restrict__ o2, float* __restrict__ cat) {
  size_t i = (size_t)blockIdx.x * 256 + threadIdx.x;
  if (i >= (size_t)N_NODES * DCATD) return;
  size_t r = i / DCATD, d = i % DCATD;
  float v;
  if (d < FD)             v = x[r * FD + d];
  else if (d < FD + HIDD) v = o1[r * HIDD + (d - FD)];
  else                    v = o2[r * HIDD + (d - FD - HIDD)];
  cat[i] = v;
}

__global__ __launch_bounds__(256)
void k_score_final(const float* __restrict__ score, const float* __restrict__ b2,
                   const float* __restrict__ center, float* __restrict__ sc, int n) {
  int d = blockIdx.x * 256 + threadIdx.x;
  if (d >= n) return;
  float s = score[d] + b2[0];
  sc[d] = 1.f / (1.f + __expf(-s)) - center[d];
}

__global__ __launch_bounds__(256)
void k_cls(const float* __restrict__ cat, const float* __restrict__ sc,
           const float* __restrict__ W, const float* __restrict__ b,
           float* __restrict__ out) {
  const int i = blockIdx.x;
  const int t = threadIdx.x;
  float acc[OUTD];
#pragma unroll
  for (int o = 0; o < OUTD; ++o) acc[o] = 0.f;
  for (int d = t; d < DCATD; d += 256) {
    float cv = cat[(size_t)i * DCATD + d] * sc[d];
#pragma unroll
    for (int o = 0; o < OUTD; ++o) acc[o] += cv * W[o * DCATD + d];
  }
  __shared__ float red[256];
  for (int o = 0; o < OUTD; ++o) {
    red[t] = acc[o]; __syncthreads();
    for (int s = 128; s > 0; s >>= 1) {
      if (t < s) red[t] += red[t + s];
      __syncthreads();
    }
    if (t == 0) out[(size_t)i * OUTD + o] = red[0] + b[o];
    __syncthreads();
  }
}

// ---------------- host orchestration ----------------
extern "C" void kernel_launch(void* const* d_in, const int* in_sizes, int n_in,
                              void* d_out, int out_size, void* d_ws, size_t ws_size,
                              hipStream_t stream) {
  const float* x      = (const float*)d_in[0];
  const int*   ei     = (const int*)d_in[1];
  const float* eattr  = (const float*)d_in[2];
  const float* gnw[2] = {(const float*)d_in[3], (const float*)d_in[6]};
  const float* gnb[2] = {(const float*)d_in[4], (const float*)d_in[7]};
  const float* gnm[2] = {(const float*)d_in[5], (const float*)d_in[8]};
  const float* hgW[2] = {(const float*)d_in[9],  (const float*)d_in[12]};
  const float* hgA[2] = {(const float*)d_in[10], (const float*)d_in[13]};
  const float* hgB[2] = {(const float*)d_in[11], (const float*)d_in[14]};
  const float* fcW[2] = {(const float*)d_in[15], (const float*)d_in[17]};
  const float* fcb[2] = {(const float*)d_in[16], (const float*)d_in[18]};
  const float* aW1 = (const float*)d_in[19];
  const float* ab1 = (const float*)d_in[20];
  const float* aW2 = (const float*)d_in[21];
  const float* ab2 = (const float*)d_in[22];
  const float* clsW = (const float*)d_in[23];
  const float* clsb = (const float*)d_in[24];
  const float* center = (const float*)d_in[25];
  const int* row = ei;
  const int* col = ei + NNZE;

  const size_t NF  = (size_t)N_NODES * FD;
  const size_t NH  = (size_t)N_NODES * HIDD;
  const size_t NC  = (size_t)N_NODES * DCATD;
  float* ws = (float*)d_ws;
  float* XN = ws;
  float* XT = XN + NF;
  float* ET = XT + NF;
  float* EF = ET + NF;
  float* H1 = EF + NF;
  float* H2 = H1 + NF;
  float* O1 = H2 + NF;
  float* O2 = O1 + NH;
  float* CAT = O2 + NH;
  float* SX = CAT + NC;
  float* SE = SX + N_NODES;
  float* ALPHA = SE + N_NODES;
  float* EV = ALPHA + NNZE;
  float* SSUM = EV + NNZE;
  float* DEGE = SSUM + N_NODES;
  float* DEGN = DEGE + N_NODES;
  unsigned* MCOL = (unsigned*)(DEGN + N_NODES);
  float* SCALE = DEGN + 2 * N_NODES;
  float* SHIFT = SCALE + FD;
  float* SCORE = SHIFT + FD;
  float* SC = SCORE + DCATD;

  const dim3 blk(256);
  auto nblk = [](size_t n) { return dim3((unsigned)((n + 255) / 256)); };

  float* Hbuf[2] = {H1, H2};
  float* Obuf[2] = {O1, O2};

  for (int l = 0; l < 2; ++l) {
    const float* Xin = (l == 0) ? x : H1;
    float* Hout = Hbuf[l];
    // GraphNorm
    k_colstats<<<dim3(FD), blk, 0, stream>>>(Xin, gnw[l], gnb[l], gnm[l], SCALE, SHIFT);
    k_gn_apply<<<nblk(NF / 4), blk, 0, stream>>>(Xin, SCALE, SHIFT, XN, NF);
    // xt = xn @ W^T, et = edge_attr @ W^T  (bf16 WMMA, f32 acc)
    k_wmma_gemm<<<dim3(FD / 64, N_NODES / 128), blk, 0, stream>>>(
        XN, hgW[l], XT, N_NODES, FD, FD, nullptr, 0, 1);
    k_wmma_gemm<<<dim3(FD / 64, N_NODES / 128), blk, 0, stream>>>(
        eattr, hgW[l], ET, N_NODES, FD, FD, nullptr, 0, 1);
    // per-node attention logits
    k_rowdot<<<dim3(N_NODES), blk, 0, stream>>>(XT, ET, hgA[l], SX, SE);
    // clear segment scratch + accumulation targets
    k_fill_u32<<<nblk(N_NODES), blk, 0, stream>>>((unsigned*)SSUM, 0u, N_NODES);
    k_fill_u32<<<nblk(N_NODES), blk, 0, stream>>>((unsigned*)DEGE, 0u, N_NODES);
    k_fill_u32<<<nblk(N_NODES), blk, 0, stream>>>((unsigned*)DEGN, 0u, N_NODES);
    k_fill_u32<<<nblk(N_NODES), blk, 0, stream>>>(MCOL, 0x007FFFFFu, N_NODES); // enc(-inf)
    k_fill_u32<<<nblk(NF), blk, 0, stream>>>((unsigned*)EF, 0u, NF);
    k_fill_u32<<<nblk(NF), blk, 0, stream>>>((unsigned*)Hout, 0u, NF);
    // segment softmax + degrees + scatters
    k_edge_alpha_max<<<nblk(NNZE), blk, 0, stream>>>(row, col, SX, SE, ALPHA, MCOL, DEGE, DEGN);
    k_edge_exp_sum<<<nblk(NNZE), blk, 0, stream>>>(col, ALPHA, MCOL, EV, SSUM);
    k_edge_scatter_efeat<<<dim3(NNZE), blk, 0, stream>>>(row, col, EV, SSUM, DEGE, XT, EF);
    k_edge_scatter_out<<<dim3(NNZE), blk, 0, stream>>>(row, col, EV, SSUM, DEGN, EF, Hout);
    // + bias, lrelu(0.01) -> h
    k_bias_lrelu<<<nblk(NF), blk, 0, stream>>>(Hout, hgB[l], FD, NF, 0.01f);
    // fc: out = lrelu(h @ fcW^T + fcb)
    k_wmma_gemm<<<dim3(HIDD / 64, N_NODES / 128), blk, 0, stream>>>(
        Hout, fcW[l], Obuf[l], N_NODES, HIDD, FD, fcb[l], 1, 1);
  }

  // cat = [x | out1 | out2]
  k_concat<<<nblk(NC), blk, 0, stream>>>(x, O1, O2, CAT);
  // fused attn GEMM + relu*W2 reduction -> score[DCAT]
  k_fill_u32<<<nblk(DCATD), blk, 0, stream>>>((unsigned*)SCORE, 0u, DCATD);
  k_attn_gemm_score<<<dim3(DCATD / 64, N_NODES / 128), blk, 0, stream>>>(aW1, CAT, ab1, aW2, SCORE);
  k_score_final<<<nblk(DCATD), blk, 0, stream>>>(SCORE, ab2, center, SC, DCATD);
  // logits = (cat * sc) @ clsW^T + clsb
  k_cls<<<dim3(N_NODES), blk, 0, stream>>>(CAT, SC, clsW, clsb, (float*)d_out);
}
